// OneDUNet_58471684768010
// MI455X (gfx1250) — matrix operand order repaired and checked by reading, not compiled, not run on hardware
//
#include <hip/hip_runtime.h>
#include <hip/hip_bf16.h>

typedef __attribute__((ext_vector_type(16))) __bf16 v16bf;
typedef __attribute__((ext_vector_type(8)))  float  v8f;

// ---------- bf16 WMMA GEMM: out[M x C] = act(A[M x K]) @ W[K x C] (+ bias) ----------
// act(x) = relu(x*scale[k] + shift[k]) applied to A elements when ACT=true
template <int K, int C, bool ACT, bool BIAS>
__global__ __launch_bounds__(128) void gemm_k(
    const float* __restrict__ A, const float* __restrict__ W,
    const float* __restrict__ bias,
    const float* __restrict__ scale, const float* __restrict__ shift,
    float* __restrict__ out, int M)
{
  constexpr int KC = (K + 31) / 32;     // K chunks of 32
  constexpr int NT = C / 16;            // 16-wide column tiles
  const int lane = threadIdx.x & 31;
  const int wave = threadIdx.x >> 5;
  const int mt   = blockIdx.x * 4 + wave;
  const int rb   = mt * 16;
  if (rb >= M) return;                  // wave-uniform exit; EXEC stays full
  const bool fulltile = (rb + 16 <= M); // true for all tiles when M % 16 == 0
  const int h   = lane >> 4;            // lane half (0/1)
  const int ln  = lane & 15;
  const int row = rb + ln;
  const int rc  = row < M ? row : M - 1;

  // A fragments: 16-bit A 16x32 layout.
  // element j<8  -> k = kc*32 + h*8 + j        (two runs of 8 consecutive floats)
  // element j>=8 -> k = kc*32 + 16 + h*8 + j-8
  v16bf afr[KC];
  if constexpr (K % 32 == 0) {
    const float* arow = A + (long)rc * K + h * 8;
#pragma unroll
    for (int kc = 0; kc < KC; ++kc) {
      float4 a0 = *(const float4*)(arow + kc * 32);
      float4 a1 = *(const float4*)(arow + kc * 32 + 4);
      float4 a2 = *(const float4*)(arow + kc * 32 + 16);
      float4 a3 = *(const float4*)(arow + kc * 32 + 20);
      float av[16] = {a0.x, a0.y, a0.z, a0.w, a1.x, a1.y, a1.z, a1.w,
                      a2.x, a2.y, a2.z, a2.w, a3.x, a3.y, a3.z, a3.w};
      if (ACT) {
        const int kb = kc * 32 + h * 8;
        float4 s0 = *(const float4*)(scale + kb);
        float4 s1 = *(const float4*)(scale + kb + 4);
        float4 s2 = *(const float4*)(scale + kb + 16);
        float4 s3 = *(const float4*)(scale + kb + 20);
        float4 t0 = *(const float4*)(shift + kb);
        float4 t1 = *(const float4*)(shift + kb + 4);
        float4 t2 = *(const float4*)(shift + kb + 16);
        float4 t3 = *(const float4*)(shift + kb + 20);
        float sv[16] = {s0.x, s0.y, s0.z, s0.w, s1.x, s1.y, s1.z, s1.w,
                        s2.x, s2.y, s2.z, s2.w, s3.x, s3.y, s3.z, s3.w};
        float tv[16] = {t0.x, t0.y, t0.z, t0.w, t1.x, t1.y, t1.z, t1.w,
                        t2.x, t2.y, t2.z, t2.w, t3.x, t3.y, t3.z, t3.w};
#pragma unroll
        for (int j = 0; j < 16; ++j) av[j] = fmaxf(fmaf(av[j], sv[j], tv[j]), 0.0f);
      }
#pragma unroll
      for (int j = 0; j < 16; ++j) afr[kc][j] = static_cast<__bf16>(av[j]);
    }
  } else {
    // ragged K (layer 0, K=5): scalar guarded loads
#pragma unroll
    for (int kc = 0; kc < KC; ++kc) {
#pragma unroll
      for (int j = 0; j < 16; ++j) {
        int k = kc * 32 + ((j < 8) ? (h * 8 + j) : (16 + h * 8 + (j - 8)));
        float v = 0.0f;
        if (k < K) {
          v = A[(long)rc * K + k];
          if (ACT) v = fmaxf(fmaf(v, scale[k], shift[k]), 0.0f);
        }
        afr[kc][j] = static_cast<__bf16>(v);
      }
    }
  }

#pragma unroll
  for (int nt = 0; nt < NT; ++nt) {
    const int col = nt * 16 + ln;
    v8f acc = {0.f, 0.f, 0.f, 0.f, 0.f, 0.f, 0.f, 0.f};
    const float* wcol = W + col;
#pragma unroll
    for (int kc = 0; kc < KC; ++kc) {
      // B fragment: 32x16, same k order as A: j<8 -> K=kc*32+h*8+j, j>=8 -> K=kc*32+16+h*8+j-8
      v16bf bfr;
#pragma unroll
      for (int j = 0; j < 16; ++j) {
        int k = kc * 32 + ((j < 8) ? (h * 8 + j) : (16 + h * 8 + (j - 8)));
        float v;
        if constexpr (K % 32 == 0) v = wcol[(long)k * C];
        else v = (k < K) ? wcol[(long)k * C] : 0.0f;
        bfr[j] = static_cast<__bf16>(v);
      }
      acc = __builtin_amdgcn_wmma_f32_16x16x32_bf16(
          false, afr[kc], false, bfr, (short)0, acc, false, false);
    }
    const float bs = BIAS ? bias[col] : 0.0f;
    float* orow = out + (long)(rb + h * 8) * C + col;   // C/D: VGPR v -> M = v + 8*half
    if (fulltile) {
#pragma unroll
      for (int v = 0; v < 8; ++v) orow[(long)v * C] = acc[v] + bs;
    } else {
#pragma unroll
      for (int v = 0; v < 8; ++v)
        if (rb + h * 8 + v < M) orow[(long)v * C] = acc[v] + bs;
    }
  }
}

// ---------- small utility kernels ----------
__global__ void zero_k(float* p, long n) {
  for (long i = (long)blockIdx.x * blockDim.x + threadIdx.x; i < n;
       i += (long)gridDim.x * blockDim.x) p[i] = 0.0f;
}
__global__ void initconst_k(float* p, float v, long n) {
  long i = (long)blockIdx.x * blockDim.x + threadIdx.x;
  if (i < n) p[i] = v;
}
__global__ void degcount_k(const int* __restrict__ dst, float* __restrict__ deg, int E) {
  int e = blockIdx.x * blockDim.x + threadIdx.x;
  if (e < E) atomicAdd(&deg[dst[e]], 1.0f);
}
__global__ void dinv_k(const float* __restrict__ deg, float* __restrict__ dinv, int N) {
  int i = blockIdx.x * blockDim.x + threadIdx.x;
  if (i < N) { float d = deg[i]; dinv[i] = d > 0.0f ? rsqrtf(d) : 0.0f; }
}

// per-channel sum / sumsq partials (blockDim must be a multiple of C; C power of 2)
__global__ __launch_bounds__(256) void colstats_k(
    const float* __restrict__ x, int M, int C,
    float* __restrict__ sum, float* __restrict__ sumsq)
{
  int c    = threadIdx.x % C;
  int sub  = threadIdx.x / C;
  int nsub = blockDim.x / C;
  float s = 0.0f, q = 0.0f;
  for (int r = blockIdx.x * nsub + sub; r < M; r += gridDim.x * nsub) {
    float v = x[(long)r * C + c]; s += v; q += v * v;
  }
  __shared__ float ls[256], lq[256];
  ls[threadIdx.x] = s; lq[threadIdx.x] = q; __syncthreads();
  for (int st = blockDim.x / 2; st >= C; st >>= 1) {
    if ((int)threadIdx.x < st) {
      ls[threadIdx.x] += ls[threadIdx.x + st];
      lq[threadIdx.x] += lq[threadIdx.x + st];
    }
    __syncthreads();
  }
  if ((int)threadIdx.x < C) {
    atomicAdd(&sum[c],   ls[threadIdx.x]);
    atomicAdd(&sumsq[c], lq[threadIdx.x]);
  }
}

__global__ void bnfin_k(const float* __restrict__ sum, const float* __restrict__ sumsq,
                        const float* __restrict__ g, const float* __restrict__ b,
                        float Minv, float* __restrict__ scale, float* __restrict__ shift, int C)
{
  int c = blockIdx.x * blockDim.x + threadIdx.x;
  if (c >= C) return;
  float mean = sum[c] * Minv;
  float var  = sumsq[c] * Minv - mean * mean;
  float inv  = rsqrtf(var + 1e-5f);
  float sc   = g[c] * inv;
  scale[c] = sc;
  shift[c] = b[c] - mean * sc;
}

// hraw[i, q*4..] = dinv[i]^2 * m[i, q*4..]  (self-loop term; also init of scatter target)
__global__ void selfloop_k(const float* __restrict__ m, const float* __restrict__ dinv,
                           float* __restrict__ hraw, int N, int C4)
{
  long idx = (long)blockIdx.x * blockDim.x + threadIdx.x;
  if (idx >= (long)N * C4) return;
  int i = (int)(idx / C4);
  float d = dinv[i]; d = d * d;
  float4 v = *(const float4*)(m + idx * 4);
  float4 o; o.x = d * v.x; o.y = d * v.y; o.z = d * v.z; o.w = d * v.w;
  *(float4*)(hraw + idx * 4) = o;
}

// hraw[col, q*4..] += dinv[row]*dinv[col] * m[row, q*4..]  over all E edges
__global__ void scatter_k(const float* __restrict__ m, const int* __restrict__ src,
                          const int* __restrict__ dst, const float* __restrict__ dinv,
                          float* __restrict__ hraw, int E, int C4)
{
  long idx = (long)blockIdx.x * blockDim.x + threadIdx.x;
  if (idx >= (long)E * C4) return;
  int e = (int)(idx / C4);
  int q = (int)(idx % C4);
  int r = src[e], t = dst[e];
  float w = dinv[r] * dinv[t];
  float4 v = *(const float4*)(m + ((long)r * C4 + q) * 4);
  float* o = hraw + ((long)t * C4 + q) * 4;
  atomicAdd(o + 0, w * v.x);
  atomicAdd(o + 1, w * v.y);
  atomicAdd(o + 2, w * v.z);
  atomicAdd(o + 3, w * v.w);
}

// pooled[g, q*4..] += relu(bn(hraw[i, q*4..])); cnt[g] += 1 (once per node)
__global__ void pool_k(const float* __restrict__ hraw, const float* __restrict__ scale,
                       const float* __restrict__ shift, const int* __restrict__ batch,
                       float* __restrict__ pooled, float* __restrict__ cnt, int N, int C4)
{
  long idx = (long)blockIdx.x * blockDim.x + threadIdx.x;
  if (idx >= (long)N * C4) return;
  int i = (int)(idx / C4);
  int q = (int)(idx % C4);
  float4 v = *(const float4*)(hraw + idx * 4);
  float4 s = *(const float4*)(scale + q * 4);
  float4 t = *(const float4*)(shift + q * 4);
  int g = batch[i];
  float* o = pooled + ((long)g * C4 + q) * 4;
  atomicAdd(o + 0, fmaxf(fmaf(v.x, s.x, t.x), 0.0f));
  atomicAdd(o + 1, fmaxf(fmaf(v.y, s.y, t.y), 0.0f));
  atomicAdd(o + 2, fmaxf(fmaf(v.z, s.z, t.z), 0.0f));
  atomicAdd(o + 3, fmaxf(fmaf(v.w, s.w, t.w), 0.0f));
  if (q == 0) atomicAdd(&cnt[g], 1.0f);
}

__global__ void pooldiv_k(const float* __restrict__ pooled, const float* __restrict__ cnt,
                          float* __restrict__ out, int B, int C)
{
  int idx = blockIdx.x * blockDim.x + threadIdx.x;
  if (idx >= B * C) return;
  int b = idx / C;
  out[idx] = pooled[idx] / fmaxf(cnt[b], 1.0f);
}

// tiny decoder GEMM: out[B x Nc] = (relu?)(in[B x K] @ W[K x Nc] + b)
__global__ void dense_k(const float* __restrict__ in, const float* __restrict__ W,
                        const float* __restrict__ b, float* __restrict__ out,
                        int B, int K, int Nc, int dorelu)
{
  int idx = blockIdx.x * blockDim.x + threadIdx.x;
  if (idx >= B * Nc) return;
  int bb = idx / Nc, n = idx % Nc;
  float acc = b[n];
#pragma unroll 4
  for (int k = 0; k < K; ++k) acc = fmaf(in[bb * K + k], W[(long)k * Nc + n], acc);
  out[idx] = dorelu ? fmaxf(acc, 0.0f) : acc;
}

// ---------- host-side per-layer driver ----------
template <int CIN, int COUT, bool FIRST>
static void run_layer(const float* Ain, const float* Wm, const float* bm,
                      const float* g1v, const float* b1v,
                      const float* Wc, const float* g2v, const float* b2v,
                      float* lin, float* m, float* hraw,
                      float* stats, float* scale, float* shift,
                      const float* dinv, const int* src, const int* dst,
                      int N, int E, hipStream_t st)
{
  int gblocks = ((N + 15) / 16 + 3) / 4;
  // Linear (+bias); BN+ReLU of previous conv fused into A load (identity for layer 0)
  gemm_k<CIN, COUT, !FIRST, true><<<gblocks, 128, 0, st>>>(Ain, Wm, bm, scale, shift, lin, N);
  // BN stats of lin -> scale/shift with g1/b1
  zero_k<<<1, 256, 0, st>>>(stats, 256);
  colstats_k<<<256, 256, 0, st>>>(lin, N, COUT, stats, stats + 128);
  bnfin_k<<<1, 128, 0, st>>>(stats, stats + 128, g1v, b1v, 1.0f / N, scale, shift, COUT);
  // Conv transform: m = relu(bn(lin)) @ Wc   (bc cancels under the following BatchNorm)
  gemm_k<COUT, COUT, true, false><<<gblocks, 128, 0, st>>>(lin, Wc, nullptr, scale, shift, m, N);
  // Message passing: self-loops init + edge scatter-add (L2-resident m)
  constexpr int C4 = COUT / 4;
  long nc = (long)N * C4;
  selfloop_k<<<(int)((nc + 255) / 256), 256, 0, st>>>(m, dinv, hraw, N, C4);
  long ec = (long)E * C4;
  scatter_k<<<(int)((ec + 255) / 256), 256, 0, st>>>(m, src, dst, dinv, hraw, E, C4);
  // BN stats of conv output -> scale/shift with g2/b2 (consumed by next stage)
  zero_k<<<1, 256, 0, st>>>(stats, 256);
  colstats_k<<<256, 256, 0, st>>>(hraw, N, COUT, stats, stats + 128);
  bnfin_k<<<1, 128, 0, st>>>(stats, stats + 128, g2v, b2v, 1.0f / N, scale, shift, COUT);
}

extern "C" void kernel_launch(void* const* d_in, const int* in_sizes, int n_in,
                              void* d_out, int out_size, void* d_ws, size_t ws_size,
                              hipStream_t stream)
{
  const float* x     = (const float*)d_in[0];
  const int*   ei    = (const int*)d_in[1];
  const int*   batch = (const int*)d_in[2];
  const int N = in_sizes[0] / 5;
  const int E = in_sizes[1] / 2;
  const int B = 16;
  const int* src = ei;
  const int* dst = ei + E;

  // params: enc{i}: Wm,bm,g1,b1,Wc,bc,g2,b2 ; dec{j}: W,b
  const float *Wm[4], *bm[4], *g1[4], *b1[4], *Wc[4], *g2[4], *b2[4];
  int p = 3;
  for (int i = 0; i < 4; ++i) {
    Wm[i] = (const float*)d_in[p++]; bm[i] = (const float*)d_in[p++];
    g1[i] = (const float*)d_in[p++]; b1[i] = (const float*)d_in[p++];
    Wc[i] = (const float*)d_in[p++]; p++; /* bc cancels under BatchNorm */
    g2[i] = (const float*)d_in[p++]; b2[i] = (const float*)d_in[p++];
  }
  const float *dW[3], *db[3];
  for (int j = 0; j < 3; ++j) { dW[j] = (const float*)d_in[p++]; db[j] = (const float*)d_in[p++]; }

  // workspace carve (256B aligned)
  char* w = (char*)d_ws;
  auto carve = [&](size_t bytes) -> float* {
    float* r = (float*)w;
    w += (bytes + 255) & ~(size_t)255;
    return r;
  };
  float* hraw   = carve((size_t)N * 128 * 4);
  float* lin    = carve((size_t)N * 128 * 4);
  float* m      = carve((size_t)N * 128 * 4);
  float* deg    = carve((size_t)N * 4);
  float* dinv   = carve((size_t)N * 4);
  float* stats  = carve(256 * 4);          // sum[128] | sumsq[128]
  float* scale  = carve(128 * 4);
  float* shift  = carve(128 * 4);
  float* pooled = carve((size_t)(B * 128 + B) * 4);  // sums | counts
  float* cnt    = pooled + B * 128;
  float* gb     = carve((size_t)B * 128 * 4);
  float* t0     = carve((size_t)B * 256 * 4);
  float* t1     = carve((size_t)B * 512 * 4);

  // GCN degree normalization (self-loop folded in as deg init = 1)
  initconst_k<<<(N + 255) / 256, 256, 0, stream>>>(deg, 1.0f, N);
  degcount_k<<<(E + 255) / 256, 256, 0, stream>>>(dst, deg, E);
  dinv_k<<<(N + 255) / 256, 256, 0, stream>>>(deg, dinv, N);

  // encoder ladder 5 -> 32 -> 64 -> 128 -> 128
  run_layer<5,   32,  true >(x,    Wm[0], bm[0], g1[0], b1[0], Wc[0], g2[0], b2[0],
                             lin, m, hraw, stats, scale, shift, dinv, src, dst, N, E, stream);
  run_layer<32,  64,  false>(hraw, Wm[1], bm[1], g1[1], b1[1], Wc[1], g2[1], b2[1],
                             lin, m, hraw, stats, scale, shift, dinv, src, dst, N, E, stream);
  run_layer<64,  128, false>(hraw, Wm[2], bm[2], g1[2], b1[2], Wc[2], g2[2], b2[2],
                             lin, m, hraw, stats, scale, shift, dinv, src, dst, N, E, stream);
  run_layer<128, 128, false>(hraw, Wm[3], bm[3], g1[3], b1[3], Wc[3], g2[3], b2[3],
                             lin, m, hraw, stats, scale, shift, dinv, src, dst, N, E, stream);

  // global mean pool (final BN+ReLU fused into pooling)
  zero_k<<<8, 256, 0, stream>>>(pooled, (long)B * 128 + B);
  long nc4 = (long)N * 32;
  pool_k<<<(int)((nc4 + 255) / 256), 256, 0, stream>>>(hraw, scale, shift, batch, pooled, cnt, N, 32);
  pooldiv_k<<<(B * 128 + 255) / 256, 256, 0, stream>>>(pooled, cnt, gb, B, 128);

  // decoder MLP
  dense_k<<<(B * 256 + 255) / 256, 256, 0, stream>>>(gb, dW[0], db[0], t0, B, 128, 256, 1);
  dense_k<<<(B * 512 + 255) / 256, 256, 0, stream>>>(t0, dW[1], db[1], t1, B, 256, 512, 1);
  dense_k<<<(B * 3750 + 255) / 256, 256, 0, stream>>>(t1, dW[2], db[2], (float*)d_out, B, 512, 3750, 0);
}
